// GCN_61083024884001
// MI455X (gfx1250) — matrix operand order repaired and checked by reading, not compiled, not run on hardware
//
#include <hip/hip_runtime.h>
#include <hip/hip_bf16.h>
#include <stdint.h>

// ---- problem constants (from reference setup_inputs) ----
#define NN   10000      // nodes
#define EE   160000     // edges
#define FIN  3703       // input features
#define KPAD 3712       // FIN padded to multiple of 64 for pipelined K-chunks
#define KC   64         // K-chunk staged per pipeline stage (2 WMMA K-steps)
#define HID  1024       // hidden dim
#define NCLS 6          // classes

typedef __attribute__((ext_vector_type(16))) __bf16 v16bf;
typedef __attribute__((ext_vector_type(8)))  float  v8f;
typedef __attribute__((ext_vector_type(4)))  unsigned int u32x4;
typedef __attribute__((ext_vector_type(8)))  int i32x8;
typedef __attribute__((ext_vector_type(4)))  int i32x4;

#if defined(__gfx1250__) && __has_builtin(__builtin_amdgcn_tensor_load_to_lds)
#define HAVE_TDM 1
#else
#define HAVE_TDM 0
#endif

__device__ __forceinline__ unsigned short f2bf(float f) {
  unsigned u = __builtin_bit_cast(unsigned, f);
  u += 0x7FFFu + ((u >> 16) & 1u);          // round-to-nearest-even
  return (unsigned short)(u >> 16);
}

// ---------------- degree / normalization ----------------
__global__ void k_init_deg(float* deg) {
  int i = blockIdx.x * blockDim.x + threadIdx.x;
  if (i < NN) deg[i] = 1.0f;                 // self loop
}

__global__ void k_deg_accum(const long long* __restrict__ ei, float* deg) {
  int e = blockIdx.x * blockDim.x + threadIdx.x;
  if (e < EE) atomicAdd(&deg[(int)ei[EE + e]], 1.0f);   // in-degree of dst
}

__global__ void k_inv_sqrt(float* deg) {
  int i = blockIdx.x * blockDim.x + threadIdx.x;
  if (i < NN) deg[i] = rsqrtf(deg[i]);       // in-place: deg -> deg^{-1/2}
}

// ---------------- fp32 -> bf16 conversion (with K padding) ----------------
__global__ void k_cvt_x(const float* __restrict__ x, unsigned short* __restrict__ Xb) {
  long long i = (long long)blockIdx.x * blockDim.x + threadIdx.x;
  if (i >= (long long)NN * KPAD) return;
  int n = (int)(i / KPAD), k = (int)(i % KPAD);
  Xb[i] = (k < FIN) ? f2bf(x[(size_t)n * FIN + k]) : (unsigned short)0;
}

// W1 stored TRANSPOSED: WbT[n][k] -> B fragments are K-contiguous in LDS
__global__ void k_cvt_wT(const float* __restrict__ W1, unsigned short* __restrict__ WbT) {
  long long i = (long long)blockIdx.x * blockDim.x + threadIdx.x;
  if (i >= (long long)HID * KPAD) return;
  int n = (int)(i / KPAD), k = (int)(i % KPAD);
  WbT[i] = (k < FIN) ? f2bf(W1[(size_t)k * HID + n]) : (unsigned short)0;
}

// ---------------- GEMM1: H1[NN,HID] = Xb(bf16) @ W1(bf16), f32 accumulate ----
// block: 256 threads = 8 wave32s; block tile 128(M) x 128(N);
// wave grid 4x2, each wave owns 32x64 = 2x4 v_wmma_f32_16x16x32_bf16 tiles,
// 16 WMMAs per pipeline stage (KC=64). Double-buffered LDS; TDM loads for
// stage s+1 overlap the WMMAs of stage s.
extern "C" __global__ __launch_bounds__(256)
void k_gemm1_wmma(const unsigned short* __restrict__ Xb,
                  const unsigned short* __restrict__ WbT,
                  float* __restrict__ H1) {
  __shared__ unsigned short As[2][128][KC];   // 2 x 16 KB
  __shared__ unsigned short Bs[2][128][KC];   // 2 x 16 KB

  const int tid  = threadIdx.x;
  const int wave = tid >> 5;
  const int lane = tid & 31;
  const int wm   = wave >> 1;                // 0..3  (32-row M sub-tile)
  const int wn   = wave & 1;                 // 0..1  (64-col N sub-tile)
  const int mBase = blockIdx.x * 128;
  const int nBase = blockIdx.y * 128;
  const int NSTEP = KPAD / KC;               // 58 stages

  v8f acc[2][4] = {};

#if HAVE_TDM
  // ---- Tensor DMA descriptors: 2D tile KC(K-contig) x 128(rows), bf16 ----
  const unsigned ldsA0 = (unsigned)(size_t)(__attribute__((address_space(3))) void*)&As[0][0][0];
  const unsigned ldsA1 = (unsigned)(size_t)(__attribute__((address_space(3))) void*)&As[1][0][0];
  const unsigned ldsB0 = (unsigned)(size_t)(__attribute__((address_space(3))) void*)&Bs[0][0][0];
  const unsigned ldsB1 = (unsigned)(size_t)(__attribute__((address_space(3))) void*)&Bs[1][0][0];
  const int rowsA = NN - mBase;              // hardware zero-fills OOB rows
  const int rowsB = HID - nBase;
  // group1: data_size=2B; tensor_dim0=KPAD; tensor_dim1=rows; tile=KCx128;
  //         tensor_dim0_stride=KPAD
  const i32x8 g1a = { 0x10000, (int)((unsigned)KPAD << 16),
                      (int)((unsigned)(rowsA & 0xffff) << 16), (int)((unsigned)KC << 16),
                      128, KPAD, 0, 0 };
  const i32x8 g1b = { 0x10000, (int)((unsigned)KPAD << 16),
                      (int)((unsigned)(rowsB & 0xffff) << 16), (int)((unsigned)KC << 16),
                      128, KPAD, 0, 0 };
  const i32x4 gz4 = {};
#if __clang_major__ >= 23
  const i32x8 gz8 = {};
#endif

  auto tdm_issue = [&](int buf, int k0) {
    const unsigned long long gaA =
        (unsigned long long)(size_t)(Xb  + (size_t)mBase * KPAD + k0);
    const unsigned long long gaB =
        (unsigned long long)(size_t)(WbT + (size_t)nBase * KPAD + k0);
    // group0: count=1 | lds_addr | global_addr[56:0] | type=2
    const u32x4 g0a = { 1u, buf ? ldsA1 : ldsA0, (unsigned)gaA,
                        (unsigned)((gaA >> 32) & 0x1ffffffu) | 0x80000000u };
    const u32x4 g0b = { 1u, buf ? ldsB1 : ldsB0, (unsigned)gaB,
                        (unsigned)((gaB >> 32) & 0x1ffffffu) | 0x80000000u };
#if __clang_major__ >= 23
    __builtin_amdgcn_tensor_load_to_lds(g0a, g1a, gz4, gz4, gz8, 0);
    __builtin_amdgcn_tensor_load_to_lds(g0b, g1b, gz4, gz4, gz8, 0);
#else
    __builtin_amdgcn_tensor_load_to_lds(g0a, g1a, gz4, gz4, 0);
    __builtin_amdgcn_tensor_load_to_lds(g0b, g1b, gz4, gz4, 0);
#endif
  };
#endif

  union Frag { uint4 q[2]; unsigned u32[8]; unsigned short u16[16]; v16bf v; };
  const int lml = lane & 15;
  const int kbA = (lane < 16) ? 0 : 8;       // A: upper half-wave holds K+8 pairs
  const int kbB = (lane < 16) ? 0 : 8;       // B: upper half-wave holds K+8 pairs

#if HAVE_TDM
  // ------------------- pipelined TDM main loop -------------------
  if (wave == 0) tdm_issue(0, 0);            // prologue: stage 0 in flight
  for (int s = 0; s < NSTEP; ++s) {
    const int cur = s & 1;
    if (wave == 0) __builtin_amdgcn_s_wait_tensorcnt(0);  // buf cur ready
    __syncthreads();   // consumers see data; buf cur^1 fully read last stage
    if (wave == 0 && s + 1 < NSTEP) tdm_issue(cur ^ 1, (s + 1) * KC);

    #pragma unroll
    for (int kk = 0; kk < KC; kk += 32) {
      Frag Af[2], Bf[4];
      #pragma unroll
      for (int ti = 0; ti < 2; ++ti) {
        const int m = wm * 32 + ti * 16 + lml;
        Af[ti].q[0] = *(const uint4*)&As[cur][m][kk + kbA];
        Af[ti].q[1] = *(const uint4*)&As[cur][m][kk + 16 + kbA];
      }
      #pragma unroll
      for (int tj = 0; tj < 4; ++tj) {
        const int n = wn * 64 + tj * 16 + lml;
        Bf[tj].q[0] = *(const uint4*)&Bs[cur][n][kk + kbB];
        Bf[tj].q[1] = *(const uint4*)&Bs[cur][n][kk + kbB + 16];
      }
      #pragma unroll
      for (int ti = 0; ti < 2; ++ti)
        #pragma unroll
        for (int tj = 0; tj < 4; ++tj)
          acc[ti][tj] = __builtin_amdgcn_wmma_f32_16x16x32_bf16(
              false, Af[ti].v, false, Bf[tj].v,
              (short)0, acc[ti][tj], false, false);
    }
    __syncthreads();   // buf cur fully read before it is refilled at stage s+2
  }
#else
  // ------------------- fallback: cooperative vector staging -------------------
  for (int s = 0; s < NSTEP; ++s) {
    const int k0 = s * KC;
    __syncthreads();
    {
      const int ar = tid >> 1;               // tile row 0..127
      const int hf = tid & 1;                // which 32-elt half of KC
      uint4 a[4] = {};
      const int grow = mBase + ar;
      if (grow < NN) {
        const uint4* p = (const uint4*)(Xb + (size_t)grow * KPAD + k0 + hf * 32);
        a[0] = p[0]; a[1] = p[1]; a[2] = p[2]; a[3] = p[3];
      }
      #pragma unroll
      for (int j = 0; j < 4; ++j) *(uint4*)&As[0][ar][hf * 32 + j * 8] = a[j];
      const uint4* q = (const uint4*)(WbT + (size_t)(nBase + ar) * KPAD + k0 + hf * 32);
      #pragma unroll
      for (int j = 0; j < 4; ++j) *(uint4*)&Bs[0][ar][hf * 32 + j * 8] = q[j];
    }
    __syncthreads();
    #pragma unroll
    for (int kk = 0; kk < KC; kk += 32) {
      Frag Af[2], Bf[4];
      #pragma unroll
      for (int ti = 0; ti < 2; ++ti) {
        const int m = wm * 32 + ti * 16 + lml;
        Af[ti].q[0] = *(const uint4*)&As[0][m][kk + kbA];
        Af[ti].q[1] = *(const uint4*)&As[0][m][kk + 16 + kbA];
      }
      #pragma unroll
      for (int tj = 0; tj < 4; ++tj) {
        const int n = wn * 64 + tj * 16 + lml;
        Bf[tj].q[0] = *(const uint4*)&Bs[0][n][kk + kbB];
        Bf[tj].q[1] = *(const uint4*)&Bs[0][n][kk + kbB + 16];
      }
      #pragma unroll
      for (int ti = 0; ti < 2; ++ti)
        #pragma unroll
        for (int tj = 0; tj < 4; ++tj)
          acc[ti][tj] = __builtin_amdgcn_wmma_f32_16x16x32_bf16(
              false, Af[ti].v, false, Bf[tj].v,
              (short)0, acc[ti][tj], false, false);
    }
  }
#endif

  // ---- epilogue: C/D layout VGPR r -> M = r + (lane>=16 ? 8 : 0), N = lane%16
  const int mhalf = (lane < 16) ? 0 : 8;
  #pragma unroll
  for (int ti = 0; ti < 2; ++ti) {
    #pragma unroll
    for (int tj = 0; tj < 4; ++tj) {
      const int col = nBase + wn * 64 + tj * 16 + lml;
      #pragma unroll
      for (int rr = 0; rr < 8; ++rr) {
        const int row = mBase + wm * 32 + ti * 16 + mhalf + rr;
        if (row < NN) H1[(size_t)row * HID + col] = acc[ti][tj][rr];
      }
    }
  }
}

// ---------------- aggregation buffers ----------------
__global__ void k_init_h1a(const float* __restrict__ b1, float* __restrict__ H1A) {
  long long i = (long long)blockIdx.x * blockDim.x + threadIdx.x;
  if (i < (long long)NN * HID) H1A[i] = b1[i & (HID - 1)];  // init with bias
}

// one block per (edge or self-loop); 256 threads sweep 1024 features
__global__ void k_scatter1(const long long* __restrict__ ei,
                           const float* __restrict__ inv,
                           const float* __restrict__ H1,
                           float* __restrict__ H1A) {
  const int e = blockIdx.x;
  int s, d;
  if (e < EE) { s = (int)ei[e]; d = (int)ei[EE + e]; }
  else        { s = e - EE; d = s; }
  const float w = inv[s] * inv[d];
  const float* __restrict__ srcRow = H1  + (size_t)s * HID;
  float*       __restrict__ dstRow = H1A + (size_t)d * HID;
  __builtin_prefetch(srcRow + threadIdx.x, 0, 0);   // global_prefetch_b8
  #pragma unroll
  for (int f = threadIdx.x; f < HID; f += 256)
    atomicAdd(&dstRow[f], srcRow[f] * w);
}

__global__ void k_relu(float* H1A) {
  long long i = (long long)blockIdx.x * blockDim.x + threadIdx.x;
  if (i < (long long)NN * HID) H1A[i] = fmaxf(H1A[i], 0.0f);
}

// ---------------- layer-2 GEMM (1024x6, tiny; plain VALU) ----------------
__global__ void k_gemm2(const float* __restrict__ Hin,
                        const float* __restrict__ W2,
                        float* __restrict__ H2) {
  int idx = blockIdx.x * blockDim.x + threadIdx.x;
  int n = idx >> 3, c = idx & 7;
  if (n >= NN) return;
  float s = 0.0f;
  if (c < NCLS) {
    const float* hrow = Hin + (size_t)n * HID;
    #pragma unroll 8
    for (int k = 0; k < HID; ++k) s = fmaf(hrow[k], W2[k * NCLS + c], s);
  }
  H2[n * 8 + c] = (c < NCLS) ? s : 0.0f;
}

__global__ void k_init_h2a(const float* __restrict__ b2, float* __restrict__ H2A) {
  int idx = blockIdx.x * blockDim.x + threadIdx.x;
  if (idx >= NN * 8) return;
  int c = idx & 7;
  H2A[idx] = (c < NCLS) ? b2[c] : 0.0f;
}

__global__ void k_scatter2(const long long* __restrict__ ei,
                           const float* __restrict__ inv,
                           const float* __restrict__ H2,
                           float* __restrict__ H2A) {
  int e = blockIdx.x * blockDim.x + threadIdx.x;
  if (e >= EE + NN) return;
  int s, d;
  if (e < EE) { s = (int)ei[e]; d = (int)ei[EE + e]; }
  else        { s = e - EE; d = s; }
  const float w = inv[s] * inv[d];
  #pragma unroll
  for (int c = 0; c < NCLS; ++c)
    atomicAdd(&H2A[d * 8 + c], H2[s * 8 + c] * w);
}

__global__ void k_log_softmax(const float* __restrict__ H2A, float* __restrict__ out) {
  int n = blockIdx.x * blockDim.x + threadIdx.x;
  if (n >= NN) return;
  float v[NCLS], m = -3.4e38f;
  #pragma unroll
  for (int c = 0; c < NCLS; ++c) { v[c] = H2A[n * 8 + c]; m = fmaxf(m, v[c]); }
  float ssum = 0.0f;
  #pragma unroll
  for (int c = 0; c < NCLS; ++c) ssum += __expf(v[c] - m);
  const float l = __logf(ssum);
  #pragma unroll
  for (int c = 0; c < NCLS; ++c) out[n * NCLS + c] = v[c] - m - l;
}

// ---------------- launcher ----------------
extern "C" void kernel_launch(void* const* d_in, const int* in_sizes, int n_in,
                              void* d_out, int out_size, void* d_ws, size_t ws_size,
                              hipStream_t stream) {
  const float*     x  = (const float*)d_in[0];
  const long long* ei = (const long long*)d_in[1];   // int64 [2,E]
  const float*     W1 = (const float*)d_in[2];
  const float*     b1 = (const float*)d_in[3];
  const float*     W2 = (const float*)d_in[4];
  const float*     b2 = (const float*)d_in[5];
  float* out = (float*)d_out;

  // workspace carve-up (all 256B-aligned chunk sizes)
  char* ws = (char*)d_ws;
  float* deg = (float*)ws;                      ws += 65536;                       // -> becomes inv_sqrt
  unsigned short* Xb  = (unsigned short*)ws;    ws += (size_t)NN * KPAD * 2;       // 74.24 MB
  unsigned short* WbT = (unsigned short*)ws;    ws += (size_t)HID * KPAD * 2;      // 7.60 MB
  float* H1  = (float*)ws;                      ws += (size_t)NN * HID * 4;        // 40.96 MB
  float* H1A = (float*)ws;                      ws += (size_t)NN * HID * 4;        // 40.96 MB
  float* H2  = (float*)ws;                      ws += (size_t)NN * 8 * 4;
  float* H2A = (float*)ws;

  const int T = 256;
  k_init_deg <<<(NN + T - 1) / T, T, 0, stream>>>(deg);
  k_deg_accum<<<(EE + T - 1) / T, T, 0, stream>>>(ei, deg);
  k_inv_sqrt <<<(NN + T - 1) / T, T, 0, stream>>>(deg);

  const long long xElems = (long long)NN * KPAD;
  const long long wElems = (long long)HID * KPAD;
  k_cvt_x <<<(unsigned)((xElems + T - 1) / T), T, 0, stream>>>(x, Xb);
  k_cvt_wT<<<(unsigned)((wElems + T - 1) / T), T, 0, stream>>>(W1, WbT);

  dim3 g1((NN + 127) / 128, HID / 128);
  k_gemm1_wmma<<<g1, T, 0, stream>>>(Xb, WbT, H1);

  const long long hElems = (long long)NN * HID;
  k_init_h1a<<<(unsigned)((hElems + T - 1) / T), T, 0, stream>>>(b1, H1A);
  k_scatter1<<<EE + NN, T, 0, stream>>>(ei, deg, H1, H1A);
  k_relu    <<<(unsigned)((hElems + T - 1) / T), T, 0, stream>>>(H1A);

  k_gemm2   <<<(NN * 8 + T - 1) / T, T, 0, stream>>>(H1A, W2, H2);
  k_init_h2a<<<(NN * 8 + T - 1) / T, T, 0, stream>>>(b2, H2A);
  k_scatter2<<<(EE + NN + T - 1) / T, T, 0, stream>>>(ei, deg, H2, H2A);
  k_log_softmax<<<(NN + T - 1) / T, T, 0, stream>>>(H2A, out);

  (void)in_sizes; (void)n_in; (void)out_size; (void)ws_size;
}
